// lrcn_neureg_bn_static2_all_views_time_indp_dec_38646115729704
// MI455X (gfx1250) — compile-verified, matrix operand order checked by reading, not run on hardware
//
#include <hip/hip_runtime.h>
#include <hip/hip_bf16.h>
#include <stdint.h>

typedef __attribute__((ext_vector_type(16))) _Float16 v16h;
typedef __attribute__((ext_vector_type(8)))  float    v8f;

#define AS1 __attribute__((address_space(1)))
#define AS3 __attribute__((address_space(3)))

#if defined(__has_builtin)
#if __has_builtin(__builtin_amdgcn_global_load_async_to_lds_b128) && \
    __has_builtin(__builtin_amdgcn_s_wait_asynccnt)
#define HAVE_ASYNC_G2L 1
#endif
#endif
#ifndef HAVE_ASYNC_G2L
#define HAVE_ASYNC_G2L 0
#endif

#if HAVE_ASYNC_G2L
typedef int v4i_ __attribute__((vector_size(16)));
// GLOBAL_LOAD_ASYNC_TO_LDS_B128: per-lane HBM -> LDS copy, tracked by ASYNCcnt.
__device__ __attribute__((always_inline)) inline void
async_g2l_b128(const void* g, void* l) {
  __builtin_amdgcn_global_load_async_to_lds_b128((AS1 v4i_*)g, (AS3 v4i_*)l, 0, 0);
}
#endif

__constant__ int c_nneur[8] = {300, 350, 400, 450, 500, 420, 380, 512};

// ---------------------------------------------------------------------------
// WMMA fragment gathers (wave32, 16x16x32 f16 layouts per CDNA5 ISA 7.12.2)
// A tile in LDS: [Mpad][32] halfs, row stride 32.
// B tile in LDS: pair-major [16][TP][2] halfs: pair kp at dword (kp*TP + p).
// ---------------------------------------------------------------------------
__device__ inline v16h load_a_frag(const _Float16* As, int mBase) {
  const int lane = threadIdx.x & 31;
  const int g = lane >> 4;
  const int m = mBase + (lane & 15);
  union { v16h v; unsigned u[8]; } f;
  const unsigned* row = (const unsigned*)(As + m * 32);
#pragma unroll
  for (int v = 0; v < 8; ++v) {
    int K = (v < 4) ? (g * 8 + 2 * v) : (16 + g * 8 + 2 * (v - 4));
    f.u[v] = row[K >> 1];
  }
  return f.v;
}

__device__ inline v16h load_b_frag(const _Float16* Bs, int pBase, int TP) {
  const int lane = threadIdx.x & 31;
  const int g = lane >> 4;
  const int col = pBase + (lane & 15);
  union { v16h v; unsigned u[8]; } f;
  const unsigned* base = (const unsigned*)Bs;
#pragma unroll
  for (int v = 0; v < 8; ++v) {
    int kp = (v < 4) ? (g * 4 + v) : (8 + g * 4 + (v - 4));
    f.u[v] = base[kp * TP + col];
  }
  return f.v;
}

// ---------------------------------------------------------------------------
// Prep kernels
// ---------------------------------------------------------------------------
__global__ void pack_in_k(const float* __restrict__ x, _Float16* __restrict__ X16) {
  int idx = blockIdx.x * 256 + threadIdx.x;           // 320 * 12544
  if (idx >= 320 * 12544) return;
  int nimg = idx / 12544, p = idx - nimg * 12544;
  int b = nimg / 40, s = nimg - b * 40;               // PRED[s] = s + 5
  X16[idx] = (_Float16)x[((size_t)b * 50 + (s + 5)) * 12544 + p];
}

__global__ void pack_w_k(const float* __restrict__ cw, int view, int COUT, int CIN,
                         int Mpad, int Kpad, _Float16* __restrict__ out) {
  int idx = blockIdx.x * 256 + threadIdx.x;
  if (idx >= Mpad * Kpad) return;
  int m = idx / Kpad, k = idx - m * Kpad;
  float v = 0.f;
  if (m < COUT && k < CIN * 25)
    v = cw[((size_t)view * COUT + m) * CIN * 25 + k];
  out[idx] = (_Float16)v;
}

__global__ void prep_bn_k(const float* __restrict__ cb, const float* __restrict__ g,
                          const float* __restrict__ be, const float* __restrict__ mn,
                          const float* __restrict__ vr, int view, int C,
                          float* __restrict__ bnp) {   // [3][32]: bias, scale, shift
  int c = threadIdx.x;
  if (c < C) {
    float s = g[view * C + c] * rsqrtf(vr[view * C + c] + 1e-5f);
    bnp[c]      = cb[view * C + c];
    bnp[32 + c] = s;
    bnp[64 + c] = be[view * C + c] - mn[view * C + c] * s;
  }
}

template <int C, int HW>
__global__ void pool2_k(const _Float16* __restrict__ in, _Float16* __restrict__ out) {
  constexpr int HO = HW / 2;
  constexpr int total = 320 * C * HO * HO;
  int idx = blockIdx.x * 256 + threadIdx.x;
  if (idx >= total) return;
  int n = idx / (C * HO * HO);
  int rem = idx - n * (C * HO * HO);
  int c = rem / (HO * HO);
  int pr = rem - c * (HO * HO);
  int y = pr / HO, x = pr - y * HO;
  const _Float16* s = in + (((size_t)n * C + c) * HW + 2 * y) * HW + 2 * x;
  float a = (float)s[0], b = (float)s[1], d = (float)s[HW], e = (float)s[HW + 1];
  out[idx] = (_Float16)fmaxf(fmaxf(a, b), fmaxf(d, e));
}

// ---------------------------------------------------------------------------
// Implicit-GEMM conv: OUT[n, co, pix] = BN(ReLU(conv + bias))
// M = Cout (padded 16), N = 128-pixel tile, K = Cin*25 (padded 32), im2col->LDS.
// ---------------------------------------------------------------------------
template <int CIN, int COUT, int HW>
__global__ void __launch_bounds__(256)
conv_wmma_k(const _Float16* __restrict__ in, const _Float16* __restrict__ Wp,
            const float* __restrict__ bnp, _Float16* __restrict__ out, int outNStride) {
  constexpr int KRAW = CIN * 25;
  constexpr int Kpad = (KRAW + 31) & ~31;
  constexpr int KT = Kpad / 32;
  constexpr int Mpad = (COUT + 15) & ~15;
  constexpr int MT = Mpad / 16;
  constexpr int HWHW = HW * HW;
  constexpr int TP = 128;

  __shared__ alignas(16) _Float16 As[Mpad * 32];
  __shared__ alignas(16) _Float16 Bs[32 * TP];

  const int t = threadIdx.x;
  const int lane = t & 31;
  const int wv = t >> 5;
  const int n = blockIdx.y;
  const int tileBase = blockIdx.x * TP;

  v8f acc[MT];
  const v8f vzero = {0, 0, 0, 0, 0, 0, 0, 0};
#pragma unroll
  for (int i = 0; i < MT; ++i) acc[i] = vzero;

  const _Float16* inN = in + (size_t)n * CIN * HWHW;
  const int kLoc = t >> 3;          // 0..31 : K row of this thread
  const int pLoc = (t & 7) << 4;    // 16-pixel run

  for (int kc = 0; kc < KT; ++kc) {
    __syncthreads();
    // ---- A tile: packed weights [Mpad][Kpad] -> LDS [Mpad][32]
    constexpr int PER = (Mpad * 32) / 256;
#pragma unroll
    for (int j = 0; j < PER; ++j) {
      int e = t * PER + j;
      As[e] = Wp[(e >> 5) * Kpad + kc * 32 + (e & 31)];
    }
    // ---- B tile: im2col gather -> pair-major LDS
    {
      int kg = kc * 32 + kLoc;
      int ci = kg / 25;
      int rem = kg - ci * 25;
      int rr = rem / 5;
      int cc = rem - rr * 5;
      bool kvalid = kg < KRAW;
      const _Float16* inC = inN + (size_t)ci * HWHW;
      _Float16* bdst = Bs + ((kLoc >> 1) * TP) * 2 + (kLoc & 1);
#pragma unroll
      for (int j = 0; j < 16; ++j) {
        int p = pLoc + j;
        int pix = tileBase + p;
        _Float16 v = (_Float16)0.f;
        if (kvalid && pix < HWHW) {
          int y = pix / HW, x = pix - y * HW;
          int iy = y + rr - 2, ix = x + cc - 2;
          if (iy >= 0 && iy < HW && ix >= 0 && ix < HW) v = inC[iy * HW + ix];
        }
        bdst[p * 2] = v;
      }
    }
    __syncthreads();
    // batched fragment loads, then back-to-back WMMAs
    v16h bf = load_b_frag(Bs, wv * 16, TP);
    v16h af[MT];
#pragma unroll
    for (int mt = 0; mt < MT; ++mt) af[mt] = load_a_frag(As, mt * 16);
#pragma unroll
    for (int mt = 0; mt < MT; ++mt)
      acc[mt] = __builtin_amdgcn_wmma_f32_16x16x32_f16(
          false, af[mt], false, bf, (short)0, acc[mt], false, false);
  }

  // ---- fused bias -> ReLU -> eval-BN epilogue
  const float* cbf = bnp;
  const float* scl = bnp + 32;
  const float* sft = bnp + 64;
  const int g = lane >> 4;
  const int pix = tileBase + wv * 16 + (lane & 15);
#pragma unroll
  for (int mt = 0; mt < MT; ++mt) {
#pragma unroll
    for (int r = 0; r < 8; ++r) {
      int co = mt * 16 + g * 8 + r;
      if (co < COUT && pix < HWHW) {
        float v = acc[mt][r] + cbf[co];
        v = fmaxf(v, 0.f);
        v = v * scl[co] + sft[co];
        out[(size_t)n * outNStride + (size_t)co * HWHW + pix] = (_Float16)v;
      }
    }
  }
}

// ---------------------------------------------------------------------------
// FC: H0 = ReLU(Xcat[320x50176] @ w0[50176x1024] + b0).
// BM=160 (2 block-rows cover M=320 -> w0 streamed only ~twice), BN=64, BK=32.
// Double-buffered: A tile streams HBM->LDS via ASYNCcnt engine while WMMAs run.
// ---------------------------------------------------------------------------
__global__ void __launch_bounds__(256)
fc_wmma_k(const _Float16* __restrict__ A, const float* __restrict__ B,
          const float* __restrict__ bias, _Float16* __restrict__ out) {
  constexpr int Kdim = 50176, Ndim = 1024, BM = 160, BN = 64, MT = 5;
  constexpr int KT = Kdim / 32;
  __shared__ alignas(16) _Float16 As[2][BM * 32];   // 2 x 10 KB
  __shared__ alignas(16) _Float16 Bs[2][32 * BN];   // 2 x  4 KB
  const int t = threadIdx.x, lane = t & 31, wv = t >> 5;
  const int wm = wv >> 2, wn = wv & 3;              // waves: 2 (M) x 4 (N)
  const int bm = blockIdx.y, bn = blockIdx.x;
  const int rowBase = bm * BM;

  v8f acc[MT];
  const v8f vzero = {0, 0, 0, 0, 0, 0, 0, 0};
#pragma unroll
  for (int i = 0; i < MT; ++i) acc[i] = vzero;

  auto stage = [&](int kc, int buf) {
    // ---- A tile: 160x32 halfs = 640 x b128
#if HAVE_ASYNC_G2L
    for (int e = t; e < BM * 4; e += 256) {
      int m = e >> 2, c4 = e & 3;
      const _Float16* gp = A + (size_t)(rowBase + m) * Kdim + kc * 32 + c4 * 8;
      async_g2l_b128(gp, &As[buf][e * 8]);
    }
#else
    for (int e = t; e < BM * 4; e += 256) {
      int m = e >> 2, c4 = e & 3;
      ((uint4*)As[buf])[e] =
          *(const uint4*)(A + (size_t)(rowBase + m) * Kdim + kc * 32 + c4 * 8);
    }
#endif
    // ---- B tile: 32x64 f32 -> f16 pairs (conversion work overlaps async A)
    int k = t >> 3, p0 = (t & 7) * 8;
    const float* src = B + (size_t)(kc * 32 + k) * Ndim + bn * BN + p0;
    float4 f0 = *(const float4*)src;
    float4 f1 = *(const float4*)(src + 4);
    _Float16* bd = Bs[buf] + ((k >> 1) * BN) * 2 + (k & 1);
    bd[(p0 + 0) * 2] = (_Float16)f0.x; bd[(p0 + 1) * 2] = (_Float16)f0.y;
    bd[(p0 + 2) * 2] = (_Float16)f0.z; bd[(p0 + 3) * 2] = (_Float16)f0.w;
    bd[(p0 + 4) * 2] = (_Float16)f1.x; bd[(p0 + 5) * 2] = (_Float16)f1.y;
    bd[(p0 + 6) * 2] = (_Float16)f1.z; bd[(p0 + 7) * 2] = (_Float16)f1.w;
  };

  stage(0, 0);
  for (int kc = 0; kc < KT; ++kc) {
    const int cur = kc & 1;
#if HAVE_ASYNC_G2L
    __builtin_amdgcn_s_wait_asynccnt(0);
#endif
    __syncthreads();
    if (kc + 1 < KT) {
      __builtin_prefetch(B + (size_t)((kc + 1) * 32 + (t >> 3)) * Ndim + bn * BN, 0, 0);
      stage(kc + 1, cur ^ 1);
    }
    v16h bf = load_b_frag(Bs[cur], wn * 16, BN);
    v16h af[MT];
#pragma unroll
    for (int mt = 0; mt < MT; ++mt) af[mt] = load_a_frag(As[cur], wm * 80 + mt * 16);
#pragma unroll
    for (int mt = 0; mt < MT; ++mt)
      acc[mt] = __builtin_amdgcn_wmma_f32_16x16x32_f16(
          false, af[mt], false, bf, (short)0, acc[mt], false, false);
  }

  const int g = lane >> 4;
  const int col = bn * BN + wn * 16 + (lane & 15);
  const float bv = bias[col];
#pragma unroll
  for (int mt = 0; mt < MT; ++mt) {
#pragma unroll
    for (int r = 0; r < 8; ++r) {
      int row = rowBase + wm * 80 + mt * 16 + g * 8 + r;
      float v = fmaxf(acc[mt][r] + bv, 0.f);
      out[(size_t)row * Ndim + col] = (_Float16)v;
    }
  }
}

// ---------------------------------------------------------------------------
// Per-session MLP layer: [40 x Kdim] @ W[sess][Kdim x 512] (+bias, ReLU).
// A tile async HBM->LDS; rows 40..47 zero-padded once (never rewritten).
// FINAL layer writes f32 to d_out with the NaN neuron-count mask.
// ---------------------------------------------------------------------------
template <int Kdim, bool FINAL>
__global__ void __launch_bounds__(256)
mlp_wmma_k(const _Float16* __restrict__ Hin, const float* __restrict__ W,
           const float* __restrict__ bias, const int* __restrict__ sessIdx,
           _Float16* __restrict__ Hout, float* __restrict__ outF) {
  constexpr int Nd = 512, BN = 128, MT = 3, KT = Kdim / 32;
  __shared__ alignas(16) _Float16 As[48 * 32];
  __shared__ alignas(16) _Float16 Bs[32 * BN];
  const int t = threadIdx.x, lane = t & 31, wv = t >> 5;
  const int b = blockIdx.y, ct = blockIdx.x;
  const int sess = sessIdx[b];
  const float* Wb = W + (size_t)sess * Kdim * Nd;

  v8f acc[MT];
  const v8f vzero = {0, 0, 0, 0, 0, 0, 0, 0};
#pragma unroll
  for (int i = 0; i < MT; ++i) acc[i] = vzero;

  if (t < 32) {                     // zero the 8 pad rows (chunks 160..191)
    uint4 z = {0, 0, 0, 0};
    ((uint4*)As)[160 + t] = z;
  }

  for (int kc = 0; kc < KT; ++kc) {
    __syncthreads();
    // ---- A tile: 40 valid rows x 32 halfs = 160 x b128
#if HAVE_ASYNC_G2L
    if (t < 160) {
      int m = t >> 2, c4 = t & 3;
      const _Float16* gp = Hin + (size_t)(b * 40 + m) * Kdim + kc * 32 + c4 * 8;
      async_g2l_b128(gp, &As[t * 8]);
    }
#else
    if (t < 160) {
      int m = t >> 2, c4 = t & 3;
      ((uint4*)As)[t] =
          *(const uint4*)(Hin + (size_t)(b * 40 + m) * Kdim + kc * 32 + c4 * 8);
    }
#endif
    // ---- B tile: 32x128 f32 -> f16 pairs
    {
      int k = t >> 3, p0 = (t & 7) * 16;
      const float* src = Wb + (size_t)(kc * 32 + k) * Nd + ct * BN + p0;
      _Float16* bd = Bs + ((k >> 1) * BN) * 2 + (k & 1);
#pragma unroll
      for (int j = 0; j < 16; j += 4) {
        float4 f = *(const float4*)(src + j);
        bd[(p0 + j + 0) * 2] = (_Float16)f.x;
        bd[(p0 + j + 1) * 2] = (_Float16)f.y;
        bd[(p0 + j + 2) * 2] = (_Float16)f.z;
        bd[(p0 + j + 3) * 2] = (_Float16)f.w;
      }
    }
#if HAVE_ASYNC_G2L
    __builtin_amdgcn_s_wait_asynccnt(0);
#endif
    __syncthreads();
    v16h bf = load_b_frag(Bs, wv * 16, BN);
    v16h af[MT];
#pragma unroll
    for (int mt = 0; mt < MT; ++mt) af[mt] = load_a_frag(As, mt * 16);
#pragma unroll
    for (int mt = 0; mt < MT; ++mt)
      acc[mt] = __builtin_amdgcn_wmma_f32_16x16x32_f16(
          false, af[mt], false, bf, (short)0, acc[mt], false, false);
  }

  const int g = lane >> 4;
  const int col = ct * BN + wv * 16 + (lane & 15);
  const float bv = bias[sess * Nd + col];
#pragma unroll
  for (int mt = 0; mt < MT; ++mt) {
#pragma unroll
    for (int r = 0; r < 8; ++r) {
      int s = mt * 16 + g * 8 + r;
      if (s < 40) {
        float v = fmaxf(acc[mt][r] + bv, 0.f);
        if constexpr (FINAL) {
          float o = (col < c_nneur[sess]) ? v : __builtin_nanf("");
          outF[((size_t)b * 40 + s) * 512 + col] = o;
        } else {
          Hout[(size_t)(b * 40 + s) * Nd + col] = (_Float16)v;
        }
      }
    }
  }
}

// ---------------------------------------------------------------------------
// Launch
// ---------------------------------------------------------------------------
extern "C" void kernel_launch(void* const* d_in, const int* in_sizes, int n_in,
                              void* d_out, int out_size, void* d_ws, size_t ws_size,
                              hipStream_t stream) {
  (void)in_sizes; (void)n_in; (void)out_size; (void)ws_size;

  const float* x_side   = (const float*)d_in[0];
  const float* x_bottom = (const float*)d_in[1];
  const int*   sess     = (const int*)d_in[2];
  const float *cw[6], *cb[6], *gg[6], *be[6], *mn[6], *vr[6];
  for (int i = 0; i < 6; ++i) {
    cw[i] = (const float*)d_in[3 + i * 6 + 0];
    cb[i] = (const float*)d_in[3 + i * 6 + 1];
    gg[i] = (const float*)d_in[3 + i * 6 + 2];
    be[i] = (const float*)d_in[3 + i * 6 + 3];
    mn[i] = (const float*)d_in[3 + i * 6 + 4];
    vr[i] = (const float*)d_in[3 + i * 6 + 5];
  }
  const float* w0 = (const float*)d_in[39];
  const float* b0 = (const float*)d_in[40];
  const float* W1 = (const float*)d_in[41];
  const float* b1 = (const float*)d_in[42];
  const float* W2 = (const float*)d_in[43];
  const float* b2 = (const float*)d_in[44];
  const float* W3 = (const float*)d_in[45];
  const float* b3 = (const float*)d_in[46];

  uintptr_t p = (uintptr_t)d_ws;
  auto carve = [&](size_t bytes) -> void* {
    p = (p + 255) & ~(uintptr_t)255;
    void* r = (void*)p;
    p += bytes;
    return r;
  };
  _Float16* X16  = (_Float16*)carve(320ull * 12544 * 2);        // one view, f16
  _Float16* bufA = (_Float16*)carve(320ull * 16 * 12544 * 2);   // 128 MB ping
  _Float16* bufB = (_Float16*)carve(320ull * 8 * 12544 * 2);    //  64 MB pong
  _Float16* Xcat = (_Float16*)carve(320ull * 50176 * 2);        // concat features
  _Float16* H0   = (_Float16*)carve(320ull * 1024 * 2);
  _Float16* H1   = (_Float16*)carve(320ull * 512 * 2);
  _Float16* H2   = (_Float16*)carve(320ull * 512 * 2);

  const int CINs[6]  = {1, 8, 8, 16, 16, 32};
  const int COUTs[6] = {8, 8, 16, 16, 32, 32};
  _Float16* Wp[2][6];
  float* bnp[2][6];
  for (int v = 0; v < 2; ++v)
    for (int i = 0; i < 6; ++i) {
      int Mpad = (COUTs[i] + 15) & ~15;
      int Kpad = (CINs[i] * 25 + 31) & ~31;
      Wp[v][i]  = (_Float16*)carve((size_t)Mpad * Kpad * 2);
      bnp[v][i] = (float*)carve(96 * 4);
    }

  // weight packing + BN folding
  for (int v = 0; v < 2; ++v)
    for (int i = 0; i < 6; ++i) {
      int Mpad = (COUTs[i] + 15) & ~15;
      int Kpad = (CINs[i] * 25 + 31) & ~31;
      int tot = Mpad * Kpad;
      pack_w_k<<<(tot + 255) / 256, 256, 0, stream>>>(cw[i], v, COUTs[i], CINs[i],
                                                      Mpad, Kpad, Wp[v][i]);
      prep_bn_k<<<1, 32, 0, stream>>>(cb[i], gg[i], be[i], mn[i], vr[i], v, COUTs[i],
                                      bnp[v][i]);
    }

  // CNN trunk per view, layer 5 writes straight into the concat matrix
  for (int v = 0; v < 2; ++v) {
    const float* xv = v ? x_bottom : x_side;
    pack_in_k<<<(320 * 12544 + 255) / 256, 256, 0, stream>>>(xv, X16);
    conv_wmma_k<1, 8, 112><<<dim3(98, 320), 256, 0, stream>>>(X16, Wp[v][0], bnp[v][0],
                                                              bufA, 8 * 12544);
    conv_wmma_k<8, 8, 112><<<dim3(98, 320), 256, 0, stream>>>(bufA, Wp[v][1], bnp[v][1],
                                                              bufB, 8 * 12544);
    conv_wmma_k<8, 16, 112><<<dim3(98, 320), 256, 0, stream>>>(bufB, Wp[v][2], bnp[v][2],
                                                               bufA, 16 * 12544);
    pool2_k<16, 112><<<(320 * 16 * 56 * 56 + 255) / 256, 256, 0, stream>>>(bufA, bufB);
    conv_wmma_k<16, 16, 56><<<dim3(25, 320), 256, 0, stream>>>(bufB, Wp[v][3], bnp[v][3],
                                                               bufA, 16 * 3136);
    conv_wmma_k<16, 32, 56><<<dim3(25, 320), 256, 0, stream>>>(bufA, Wp[v][4], bnp[v][4],
                                                               bufB, 32 * 3136);
    pool2_k<32, 56><<<(320 * 32 * 28 * 28 + 255) / 256, 256, 0, stream>>>(bufB, bufA);
    conv_wmma_k<32, 32, 28><<<dim3(7, 320), 256, 0, stream>>>(bufA, Wp[v][5], bnp[v][5],
                                                              Xcat + v * 25088, 50176);
  }

  // big FC (w0 streamed twice thanks to BM=160), then per-session MLP
  fc_wmma_k<<<dim3(16, 2), 256, 0, stream>>>(Xcat, w0, b0, H0);
  mlp_wmma_k<1024, false><<<dim3(4, 8), 256, 0, stream>>>(H0, W1, b1, sess, H1, nullptr);
  mlp_wmma_k<512, false><<<dim3(4, 8), 256, 0, stream>>>(H1, W2, b2, sess, H2, nullptr);
  mlp_wmma_k<512, true><<<dim3(4, 8), 256, 0, stream>>>(H2, W3, b3, sess, nullptr,
                                                        (float*)d_out);
}
